// GraphWaveletNeuralNetwork_13383118094871
// MI455X (gfx1250) — matrix operand order repaired and checked by reading, not compiled, not run on hardware
//
#include <hip/hip_runtime.h>

// ---------------------------------------------------------------------------
// GWNN forward for MI455X (gfx1250, wave32, WMMA).
// C[8192x128] = A_f32[8192xK] @ B_bf16[Kx128], f32 accumulate via
// v_wmma_f32_16x16x32_bf16, optional row-scale (diagonal filter) + ReLU.
// 128 workgroups (64-row tiles) x 8 waves (4 row-strips x 2 col-halves);
// global->register->LDS software pipeline + GL2 prefetch hides HBM latency.
// ---------------------------------------------------------------------------

#define NROWS 8192
#define NCOLS 128
#define TM    64          // block row tile

typedef __attribute__((ext_vector_type(16))) __bf16 bf16x16;
typedef __attribute__((ext_vector_type(8)))  float  f32x8;

union FragB16 {              // 16 bf16 = 8 dwords, WMMA A/B operand
    bf16x16  v;
    unsigned u[8];
};
union Pack2 {                // two bf16 packed in a dword
    __bf16   h[2];
    unsigned u;
};
union Pack8 {                // 8 bf16 in a 16-byte vector
    uint4  q;
    __bf16 h[8];
};

__global__ void cvt_f32_bf16(const float* __restrict__ in,
                             __bf16* __restrict__ out, int n) {
    int i = blockIdx.x * blockDim.x + threadIdx.x;
    if (i < n) out[i] = (__bf16)in[i];
}

// Block: 256 threads = 8 waves. Block tile: 64 rows x 128 cols.
// Wave w: row strip (w>>1)*16, column half (w&1)*64 -> 4 accum tiles 16x16.
__global__ __launch_bounds__(256)
void gwnn_gemm_bf16(const float* __restrict__ A,      // [NROWS x K] f32 row-major
                    const __bf16* __restrict__ B,     // [K x 128]  bf16 row-major
                    int K,
                    const float* __restrict__ rowscale, // optional [NROWS]
                    int do_relu,
                    float*  __restrict__ outF,        // optional [NROWS x 128]
                    __bf16* __restrict__ outB)        // optional [NROWS x 128]
{
    // K-minor layouts, stride padded to 36 bf16 (18 dwords) to spread banks.
    __shared__ __align__(16) __bf16 lds_a [TM  * 36]; // [row][k]  (A block)
    __shared__ __align__(16) __bf16 lds_bt[128 * 36]; // [col][k]  (B block, transposed)

    const int tid  = threadIdx.x;
    const int lane = tid & 31;
    const int wave = tid >> 5;            // 0..7
    const int wr   = wave >> 1;           // row strip 0..3
    const int wc   = wave & 1;            // column half 0..1
    const int half = lane >> 4;           // lane 0-15 vs 16-31 (WMMA K-half)
    const int l16  = lane & 15;
    const int m0   = blockIdx.x * TM;     // block row base

    unsigned* __restrict__ ldsAu = (unsigned*)lds_a;
    unsigned* __restrict__ ldsBu = (unsigned*)lds_bt;

    f32x8 acc[4] = {};                    // 4 column tiles x 8 VGPRs

    // Software-pipeline registers: next tile's global data in flight while
    // the current tile computes out of LDS.
    float4 ra[2];                         // A: 64 rows x 32 K (f32)
    uint4  rb[2];                         // B: 32 K x 128 cols (bf16)

    auto load_tile = [&](int k0) {
        #pragma unroll
        for (int i = 0; i < 2; ++i) {
            int c    = tid + i * 256;     // 512 float4 chunks
            int row  = c >> 3;            // 0..63
            int colv = c & 7;             // float4 within 32-wide K strip
            const float* ap = A + (size_t)(m0 + row) * K + k0 + colv * 4;
            if (k0 + 256 < K)             // stream-ahead 8 tiles into GL2
                __builtin_prefetch(ap + 256, 0, 1);
            ra[i] = *(const float4*)ap;
        }
        #pragma unroll
        for (int i = 0; i < 2; ++i) {
            int c    = tid + i * 256;     // 512 chunks of 8 bf16
            int row  = c >> 4;            // k row 0..31
            int col8 = (c & 15) * 8;
            rb[i] = *(const uint4*)(B + (size_t)(k0 + row) * NCOLS + col8);
        }
    };

    auto store_tile = [&]() {
        #pragma unroll
        for (int i = 0; i < 2; ++i) {
            int c    = tid + i * 256;
            int row  = c >> 3;
            int colv = c & 7;
            Pack2 p0, p1;
            p0.h[0] = (__bf16)ra[i].x; p0.h[1] = (__bf16)ra[i].y;
            p1.h[0] = (__bf16)ra[i].z; p1.h[1] = (__bf16)ra[i].w;
            ldsAu[row * 18 + colv * 2]     = p0.u;
            ldsAu[row * 18 + colv * 2 + 1] = p1.u;
        }
        #pragma unroll
        for (int i = 0; i < 2; ++i) {
            int c    = tid + i * 256;
            int row  = c >> 4;
            int col8 = (c & 15) * 8;
            Pack8 p;
            p.q = rb[i];
            #pragma unroll
            for (int e = 0; e < 8; ++e)   // transpose into [col][k]
                lds_bt[(col8 + e) * 36 + row] = p.h[e];
        }
    };

    load_tile(0);                         // prologue: first tile in flight

    for (int k0 = 0; k0 < K; k0 += 32) {
        __syncthreads();                  // LDS consumers of prev tile done
        store_tile();                     // regs (waits on loads) -> LDS
        __syncthreads();                  // LDS tile visible to all waves
        if (k0 + 32 < K)
            load_tile(k0 + 32);           // issue next tile's global loads now

        // ---- A fragment: 16x32 bf16, §7.12.2 layout (K pairs per VGPR) ----
        FragB16 afr;
        const int mloc = wr * 16 + l16;
        #pragma unroll
        for (int v = 0; v < 8; ++v) {
            int kk = (v < 4) ? (half * 8 + v * 2)
                             : (16 + half * 8 + (v - 4) * 2);
            afr.u[v] = ldsAu[mloc * 18 + (kk >> 1)];
        }

        // ---- 4 column tiles: B fragment (32x16) + WMMA ----
        #pragma unroll
        for (int j = 0; j < 4; ++j) {
            FragB16 bfr;
            const int n = wc * 64 + j * 16 + l16;
            #pragma unroll
            for (int v = 0; v < 8; ++v)   // same K order as A: K 8h..8h+7, then 16+8h..16+8h+7
                bfr.u[v] = ldsBu[n * 18 + ((v < 4) ? (half * 4 + v) : (8 + half * 4 + (v - 4)))];
            acc[j] = __builtin_amdgcn_wmma_f32_16x16x32_bf16(
                false, afr.v, false, bfr.v, (short)0, acc[j], false, false);
        }
    }

    // ---- epilogue: diag-filter scale, ReLU, store f32 and/or bf16 ----
    #pragma unroll
    for (int j = 0; j < 4; ++j) {
        const int n = wc * 64 + j * 16 + l16;
        #pragma unroll
        for (int r = 0; r < 8; ++r) {
            const int m = m0 + wr * 16 + half * 8 + r;    // C/D VGPR layout
            float v = acc[j][r];
            if (rowscale) v *= rowscale[m];
            if (do_relu)  v = v > 0.0f ? v : 0.0f;
            if (outF) outF[(size_t)m * NCOLS + n] = v;
            if (outB) outB[(size_t)m * NCOLS + n] = (__bf16)v;
        }
    }
}

extern "C" void kernel_launch(void* const* d_in, const int* in_sizes, int n_in,
                              void* d_out, int out_size, void* d_ws, size_t ws_size,
                              hipStream_t stream)
{
    const float* input  = (const float*)d_in[0];   // [8192 x 128]
    const float* wav    = (const float*)d_in[1];   // [8192 x 8192]
    const float* wavinv = (const float*)d_in[2];   // [8192 x 8192]
    const float* W1     = (const float*)d_in[3];   // [128 x 128]
    const float* f1     = (const float*)d_in[4];   // [8192]
    const float* W2     = (const float*)d_in[5];   // [128 x 128]
    const float* f2     = (const float*)d_in[6];   // [8192]
    float* out = (float*)d_out;                    // [8192 x 128]

    char* ws = (char*)d_ws;
    __bf16* W1bf = (__bf16*)ws;                    //  32 KB
    __bf16* W2bf = W1bf + 128 * 128;               //  32 KB
    __bf16* XWbf = W2bf + 128 * 128;               //   2 MB: X @ W (bf16)
    __bf16* Spec = XWbf + NROWS * NCOLS;           //   2 MB: f * (Winv @ XW) (bf16)
    float*  H1   = (float*)(Spec + NROWS * NCOLS); //   4 MB: relu(layer1) (f32)

    const int GB = NROWS / TM;                     // 128 blocks

    cvt_f32_bf16<<<64, 256, 0, stream>>>(W1, W1bf, 128 * 128);
    cvt_f32_bf16<<<64, 256, 0, stream>>>(W2, W2bf, 128 * 128);

    // ---- layer 1 ----
    gwnn_gemm_bf16<<<GB, 256, 0, stream>>>(input,  W1bf, 128,  nullptr, 0, nullptr, XWbf);
    gwnn_gemm_bf16<<<GB, 256, 0, stream>>>(wavinv, XWbf, 8192, f1,      0, nullptr, Spec);
    gwnn_gemm_bf16<<<GB, 256, 0, stream>>>(wav,    Spec, 8192, nullptr, 1, H1,      nullptr);
    // ---- layer 2 ----
    gwnn_gemm_bf16<<<GB, 256, 0, stream>>>(H1,     W2bf, 128,  nullptr, 0, nullptr, XWbf);
    gwnn_gemm_bf16<<<GB, 256, 0, stream>>>(wavinv, XWbf, 8192, f2,      0, nullptr, Spec);
    gwnn_gemm_bf16<<<GB, 256, 0, stream>>>(wav,    Spec, 8192, nullptr, 0, out,     nullptr);
}
